// AttentionHeadRankThree_41197326303949
// MI455X (gfx1250) — matrix operand order repaired and checked by reading, not compiled, not run on hardware
//
#include <hip/hip_runtime.h>
#include <hip/hip_bf16.h>

#define SEQ   4096
#define DIN   256
#define DOUT  64
#define BATCH 4

typedef __attribute__((ext_vector_type(16))) __bf16 v16bf;
typedef __attribute__((ext_vector_type(8)))  __bf16 v8bf;
typedef __attribute__((ext_vector_type(8)))  float  v8f;

__device__ __forceinline__ v8f wmma_bf16(v16bf a, v16bf b, v8f c) {
    return __builtin_amdgcn_wmma_f32_16x16x32_bf16(false, a, false, b, (short)0, c, false, false);
}

__device__ __forceinline__ float swap16f(float x) {
    // ds_swizzle SWAPX16: xor=0x10, and=0x1f -> offset 0x401f
    return __int_as_float(__builtin_amdgcn_ds_swizzle(__float_as_int(x), 0x401f));
}

__device__ __forceinline__ void pack8f(v16bf& dst, int base, float4 a, float4 b) {
    dst[base+0]=(__bf16)a.x; dst[base+1]=(__bf16)a.y; dst[base+2]=(__bf16)a.z; dst[base+3]=(__bf16)a.w;
    dst[base+4]=(__bf16)b.x; dst[base+5]=(__bf16)b.y; dst[base+6]=(__bf16)b.z; dst[base+7]=(__bf16)b.w;
}

// ---------------------------------------------------------------------------
// Kernel 1: QKV projection.  X[M=B*S, 256] (f32) x W[256,64] (f32) -> bf16.
// W is cooperatively staged ONCE into LDS, converted to bf16 and pre-swizzled
// into the WMMA B-operand layout; the inner loop is then ds_load_b128 + WMMA.
// blockIdx.y: 0 = K (row-major), 1 = V (stored transposed [B][64][S]), 2 = Q.
// ---------------------------------------------------------------------------
__global__ __launch_bounds__(256) void AH3_qkv_proj(
    const float* __restrict__ Xk, const float* __restrict__ Xv, const float* __restrict__ Xq,
    const float* __restrict__ Wk, const float* __restrict__ Wv, const float* __restrict__ Wq,
    __bf16* __restrict__ Qb, __bf16* __restrict__ Kb, __bf16* __restrict__ Vt)
{
    // [kc-chunk][n-tile][lane][element] = exact B-operand register image (32 KB)
    __shared__ __attribute__((aligned(32))) __bf16 s_w[8][4][32][16];

    const int which = blockIdx.y;                  // 0=K 1=V 2=Q
    const float* X = (which == 0) ? Xk : ((which == 1) ? Xv : Xq);
    const float* W = (which == 0) ? Wk : ((which == 1) ? Wv : Wq);

    const int tid  = threadIdx.x;
    const int lane = tid & 31;
    const int wave = tid >> 5;
    const int h    = lane >> 4;                    // lane half
    const int l16  = lane & 15;
    const int m0   = blockIdx.x * 128 + wave * 16; // tile row base

    // ---- cooperative, coalesced stage of W (256x64 f32 -> bf16, swizzled) ----
#pragma unroll
    for (int j = 0; j < 64; j++) {
        int i     = j * 256 + tid;                 // consecutive tids -> consecutive addrs
        int k     = i >> 6;                        // W row   (0..255)
        int n     = i & 63;                        // W col   (0..63)
        int kcid  = k >> 5;
        int kk    = k & 31;
        int bh    = kk >> 4;                       // which lane-half holds this K
        int e     = kk & 15;                       // element slot within the lane
        int nt    = n >> 4;
        int bl    = (bh << 4) | (n & 15);          // destination lane
        s_w[kcid][nt][bl][e] = (__bf16)W[i];
    }
    __syncthreads();

    const float* xrow = X + (size_t)(m0 + l16) * DIN;  // A: lane holds row M=l16
    v8f acc[4] = {};

    for (int kc = 0; kc < 8; kc++) {
        // A operand 16x32 bf16: element e -> K = kc*32 + h*16 + e
        v16bf A;
        {
            const float4* p0 = (const float4*)(xrow + kc * 32 + h * 16);
            const float4* p1 = (const float4*)(xrow + kc * 32 + h * 16 + 8);
            pack8f(A, 0, p0[0], p0[1]);
            pack8f(A, 8, p1[0], p1[1]);
        }
#pragma unroll
        for (int nt = 0; nt < 4; nt++) {
            v16bf Bv = *(const v16bf*)(&s_w[kc][nt][lane][0]);
            acc[nt] = wmma_bf16(A, Bv, acc[nt]);
        }
    }

    // C layout: lane l16 = column N%16, half h selects rows, VGPR r -> M = h*8 + r
    if (which == 1) {
        // V transposed: Vt[b][d][s]
#pragma unroll
        for (int nt = 0; nt < 4; nt++) {
            int d = nt * 16 + l16;
#pragma unroll
            for (int r = 0; r < 8; r++) {
                int m = m0 + h * 8 + r;
                int b = m >> 12;           // S = 4096
                int s = m & (SEQ - 1);
                Vt[((size_t)b * DOUT + d) * SEQ + s] = (__bf16)acc[nt][r];
            }
        }
    } else {
        __bf16* out = (which == 0) ? Kb : Qb;
#pragma unroll
        for (int nt = 0; nt < 4; nt++) {
#pragma unroll
            for (int r = 0; r < 8; r++) {
                int m = m0 + h * 8 + r;
                out[(size_t)m * DOUT + nt * 16 + l16] = (__bf16)acc[nt][r];
            }
        }
    }
}

// ---------------------------------------------------------------------------
// Kernel 2: causal flash attention.  Each wave owns 16 query rows (d=64),
// iterates 64-key tiles up to the diagonal.  Waves are independent (private
// LDS slices, no barriers).  Next key tile is prefetched (global_prefetch_b8)
// while the current tile's WMMAs/softmax run.
// ---------------------------------------------------------------------------
__global__ __launch_bounds__(256) void AH3_flash(
    const __bf16* __restrict__ Qb, const __bf16* __restrict__ Kb,
    const __bf16* __restrict__ Vt, float* __restrict__ Out)
{
    __shared__ float  s_sc[8][16][64];   // raw scores (f32)
    __shared__ __bf16 s_p [8][16][64];   // exp(scores) as bf16 (A operand for PV)
    __shared__ float  s_rs[8][16];       // per-row rescale / final 1/l

    const int lane = threadIdx.x & 31;
    const int wave = threadIdx.x >> 5;
    const int h    = lane >> 4;
    const int l16  = lane & 15;
    const int b    = blockIdx.y;
    const int qlo  = blockIdx.x * 128 + wave * 16;

    const __bf16* Qp = Qb + (size_t)b * SEQ * DOUT;
    const __bf16* Kp = Kb + (size_t)b * SEQ * DOUT;
    const __bf16* Vp = Vt + (size_t)b * DOUT * SEQ;

    // Q A-operand, 2 contraction chunks over d=64 (loaded once)
    v16bf qa[2];
    {
        const __bf16* qrow = Qp + (size_t)(qlo + l16) * DOUT;
#pragma unroll
        for (int c = 0; c < 2; c++) {
            v8bf lo = *(const v8bf*)(qrow + 32 * c + h * 16);
            v8bf hi = *(const v8bf*)(qrow + 32 * c + h * 16 + 8);
#pragma unroll
            for (int e = 0; e < 8; e++) { qa[c][e] = lo[e]; qa[c][8 + e] = hi[e]; }
        }
    }

    v8f   o[4] = {};
    float m_i  = -INFINITY;
    float l_i  = 0.0f;
    const int kmax = qlo + 15;

    for (int kb = 0; kb <= kmax; kb += 64) {   // wave-uniform bound: EXEC stays full
        // ---- prefetch next key tile (K rows + V^T segments) into caches ----
        if (kb + 64 <= kmax) {                 // wave-uniform branch
            const __bf16* kn = Kp + (size_t)(kb + 64 + lane * 2) * DOUT;      // 2 rows/lane (128B lines)
            __builtin_prefetch(kn, 0, 3);
            const __bf16* vn = Vp + (size_t)(lane * 2) * SEQ + (kb + 64);
            __builtin_prefetch(vn, 0, 3);
            __builtin_prefetch(vn + SEQ, 0, 3);
        }

        // ---- scores = Q K^T / 64, causal mask, to LDS ----
#pragma unroll
        for (int nt = 0; nt < 4; nt++) {
            const int n = kb + nt * 16 + l16;          // key index = B column
            const __bf16* krow = Kp + (size_t)n * DOUT;
            v16bf b0 = *(const v16bf*)(krow + h * 16);        // d = h*16+e
            v16bf b1 = *(const v16bf*)(krow + 32 + h * 16);   // d = 32+h*16+e
            v8f c8 = {};
            c8 = wmma_bf16(qa[0], b0, c8);
            c8 = wmma_bf16(qa[1], b1, c8);
#pragma unroll
            for (int r = 0; r < 8; r++) {
                int   mrow = h * 8 + r;
                float v    = c8[r] * 0.015625f;        // 1/sqrt(4096)
                if (n > qlo + mrow) v = -INFINITY;     // causal
                s_sc[wave][mrow][nt * 16 + l16] = v;
            }
        }
        asm volatile("s_wait_dscnt 0" ::: "memory");

        // ---- online softmax: lane owns row l16, cols h*32..h*32+31 ----
        const float* srow = &s_sc[wave][l16][h * 32];
        float tmax = -INFINITY;
#pragma unroll
        for (int j = 0; j < 32; j++) tmax = fmaxf(tmax, srow[j]);
        tmax = fmaxf(tmax, swap16f(tmax));
        float m_new = fmaxf(m_i, tmax);
        float scale = __expf(m_i - m_new);             // 0 on first tile (m_i = -inf)
        float psum  = 0.0f;
        __bf16* prow = &s_p[wave][l16][h * 32];
#pragma unroll
        for (int j = 0; j < 32; j++) {
            float p = __expf(srow[j] - m_new);
            prow[j] = (__bf16)p;
            psum += p;
        }
        psum += swap16f(psum);
        l_i = l_i * scale + psum;
        m_i = m_new;
        s_rs[wave][l16] = scale;
        asm volatile("s_wait_dscnt 0" ::: "memory");

        // ---- rescale O accumulators (per-row factor via LDS) ----
#pragma unroll
        for (int r = 0; r < 8; r++) {
            float sc = s_rs[wave][h * 8 + r];
#pragma unroll
            for (int nt = 0; nt < 4; nt++) o[nt][r] *= sc;
        }

        // ---- O += P V : A from LDS (bf16), B from transposed V (contiguous) ----
        v16bf pa[2];
#pragma unroll
        for (int c = 0; c < 2; c++) {
            v8bf lo = *(const v8bf*)(&s_p[wave][l16][32 * c + h * 16]);
            v8bf hi = *(const v8bf*)(&s_p[wave][l16][32 * c + h * 16 + 8]);
#pragma unroll
            for (int e = 0; e < 8; e++) { pa[c][e] = lo[e]; pa[c][8 + e] = hi[e]; }
        }
#pragma unroll
        for (int nt = 0; nt < 4; nt++) {
            const __bf16* vrow = Vp + (size_t)(nt * 16 + l16) * SEQ + kb;
            v16bf b0 = *(const v16bf*)(vrow + h * 16);        // key = kb+h*16+e
            v16bf b1 = *(const v16bf*)(vrow + 32 + h * 16);
            o[nt] = wmma_bf16(pa[0], b0, o[nt]);
            o[nt] = wmma_bf16(pa[1], b1, o[nt]);
        }
    }

    // ---- normalize by 1/l and store f32 output ----
    s_rs[wave][l16] = 1.0f / l_i;
    asm volatile("s_wait_dscnt 0" ::: "memory");
    float* obase = Out + ((size_t)b * SEQ + qlo) * DOUT;
#pragma unroll
    for (int r = 0; r < 8; r++) {
        float inv  = s_rs[wave][h * 8 + r];
        int   mrow = h * 8 + r;
#pragma unroll
        for (int nt = 0; nt < 4; nt++) {
            obase[(size_t)mrow * DOUT + nt * 16 + l16] = o[nt][r] * inv;
        }
    }
}

// ---------------------------------------------------------------------------
extern "C" void kernel_launch(void* const* d_in, const int* in_sizes, int n_in,
                              void* d_out, int out_size, void* d_ws, size_t ws_size,
                              hipStream_t stream)
{
    const float* Xk = (const float*)d_in[0];
    const float* Xv = (const float*)d_in[1];
    const float* Xq = (const float*)d_in[2];
    const float* Wk = (const float*)d_in[3];
    const float* Wv = (const float*)d_in[4];
    const float* Wq = (const float*)d_in[5];

    const size_t elems = (size_t)BATCH * SEQ * DOUT;  // 1 Mi elements each
    __bf16* Qb = (__bf16*)d_ws;
    __bf16* Kb = Qb + elems;
    __bf16* Vt = Kb + elems;                          // transposed [B][64][S]

    dim3 g1((BATCH * SEQ) / 128, 3);
    AH3_qkv_proj<<<g1, 256, 0, stream>>>(Xk, Xv, Xq, Wk, Wv, Wq, Qb, Kb, Vt);

    dim3 g2(SEQ / 128, BATCH);
    AH3_flash<<<g2, 256, 0, stream>>>(Qb, Kb, Vt, (float*)d_out);
}
